// MambaBlock_31997506356066
// MI455X (gfx1250) — matrix operand order repaired and checked by reading, not compiled, not run on hardware
//
#include <hip/hip_runtime.h>
#include <hip/hip_bf16.h>
#include <cstdint>

// ---------------------------------------------------------------------------
// Types for CDNA5 WMMA (wave32). 16x16x32 bf16 WMMA: A/B are v16bf (8 VGPRs),
// C/D are v8f (8 VGPRs).
// ---------------------------------------------------------------------------
typedef __attribute__((ext_vector_type(16))) __bf16       v16bf;
typedef __attribute__((ext_vector_type(8)))  float        v8f;
typedef __attribute__((ext_vector_type(4)))  unsigned int u32x4;

union FragBF {
    u32x4 q[2];   // two 16-byte chunks (8 bf16 each)
    v16bf v;
};

// One K-step worth of operands for a 32x64 tile: 2 A frags + 4 B frags.
struct FragSet {
    FragBF a0, a1, b0, b1, b2, b3;
};

__device__ __forceinline__ unsigned short f32_to_bf16_rne(float f) {
    unsigned int u = __float_as_uint(f);
    unsigned int r = u + 0x7FFFu + ((u >> 16) & 1u);
    return (unsigned short)(r >> 16);
}

// ---------------------------------------------------------------------------
// fp32 -> bf16 (round-to-nearest-even) bulk conversion
// ---------------------------------------------------------------------------
__global__ void k_f32_to_bf16(const float* __restrict__ in,
                              unsigned short* __restrict__ out, int n) {
    int i = blockIdx.x * blockDim.x + threadIdx.x;
    if (i < n) out[i] = f32_to_bf16_rne(in[i]);
}

// ---------------------------------------------------------------------------
// NT GEMM via v_wmma_f32_16x16x32_bf16, double-buffered fragment sets.
//   C[m,n] = sum_k A[m,k] * B[n,k]  (+ bias[n] if bias != nullptr)
// A: M x K bf16 row-major, B: N x K bf16 row-major, C: M x N f32 row-major.
// K is a template parameter so the 6 row base pointers collapse to 2 base
// pointers + compile-time byte offsets (folded into the load IOFFSET).
// The pipelined K-loop is forced to unroll-1 so only two fragment sets are
// ever live (64 acc + 96 frag VGPRs) -- no accumulator spills.
//
// One wave computes a 32x64 tile: 2 A fragments x 4 B fragments -> 8 WMMAs
// per K-step of 32 (12 b128 loads), loads pipelined one set ahead.
//
// Fragment layouts used here (16-bit, wave32; contiguous K, not the ISA table order):
//   A 16x32: lane r=l&15 -> row m0+r; h=l>>4 selects K halves:
//            elems 0..15 = K k0+16h .. k0+16h+15 (contig)
//   B 32x16: lane r -> col n; elems 0..15 = K k0+16h .. k0+16h+15 (contig)
//   C/D 16x16: elem j -> row m0 + 8h + j, col n0 + r
// ---------------------------------------------------------------------------
template <int K>
__device__ __forceinline__ void load_set(FragSet& s,
                                         const unsigned short* __restrict__ a0,
                                         const unsigned short* __restrict__ b0,
                                         int k0, int h) {
    const int ka = k0 + h * 16;
    s.a0.q[0] = *(const u32x4*)(a0 + ka);
    s.a0.q[1] = *(const u32x4*)(a0 + ka + 8);
    s.a1.q[0] = *(const u32x4*)(a0 + 16 * K + ka);
    s.a1.q[1] = *(const u32x4*)(a0 + 16 * K + ka + 8);
    const int kb = k0 + h * 16;
    s.b0.q[0] = *(const u32x4*)(b0 + kb);
    s.b0.q[1] = *(const u32x4*)(b0 + kb + 8);
    s.b1.q[0] = *(const u32x4*)(b0 + 16 * K + kb);
    s.b1.q[1] = *(const u32x4*)(b0 + 16 * K + kb + 8);
    s.b2.q[0] = *(const u32x4*)(b0 + 32 * K + kb);
    s.b2.q[1] = *(const u32x4*)(b0 + 32 * K + kb + 8);
    s.b3.q[0] = *(const u32x4*)(b0 + 48 * K + kb);
    s.b3.q[1] = *(const u32x4*)(b0 + 48 * K + kb + 8);
}

__device__ __forceinline__ void mma_set(const FragSet& s, v8f acc[2][4]) {
    acc[0][0] = __builtin_amdgcn_wmma_f32_16x16x32_bf16(
        false, s.a0.v, false, s.b0.v, (short)0, acc[0][0], false, false);
    acc[1][0] = __builtin_amdgcn_wmma_f32_16x16x32_bf16(
        false, s.a1.v, false, s.b0.v, (short)0, acc[1][0], false, false);
    acc[0][1] = __builtin_amdgcn_wmma_f32_16x16x32_bf16(
        false, s.a0.v, false, s.b1.v, (short)0, acc[0][1], false, false);
    acc[1][1] = __builtin_amdgcn_wmma_f32_16x16x32_bf16(
        false, s.a1.v, false, s.b1.v, (short)0, acc[1][1], false, false);
    acc[0][2] = __builtin_amdgcn_wmma_f32_16x16x32_bf16(
        false, s.a0.v, false, s.b2.v, (short)0, acc[0][2], false, false);
    acc[1][2] = __builtin_amdgcn_wmma_f32_16x16x32_bf16(
        false, s.a1.v, false, s.b2.v, (short)0, acc[1][2], false, false);
    acc[0][3] = __builtin_amdgcn_wmma_f32_16x16x32_bf16(
        false, s.a0.v, false, s.b3.v, (short)0, acc[0][3], false, false);
    acc[1][3] = __builtin_amdgcn_wmma_f32_16x16x32_bf16(
        false, s.a1.v, false, s.b3.v, (short)0, acc[1][3], false, false);
}

template <int K>
__global__ void __launch_bounds__(128)
k_wmma_gemm_nt(const unsigned short* __restrict__ A,
               const unsigned short* __restrict__ B,
               float* __restrict__ C,
               const float* __restrict__ bias,
               int M, int N) {
    const int wave   = blockIdx.x * (blockDim.x >> 5) + (threadIdx.x >> 5);
    const int lane   = threadIdx.x & 31;
    const int tilesN = N >> 6;                 // 64-wide tiles in N
    const int tm     = wave / tilesN;
    const int tn     = wave - tm * tilesN;
    const int m0     = tm * 32;                // 32-tall tiles in M
    const int n0     = tn * 64;
    if (m0 >= M) return;

    const int r = lane & 15;
    const int h = lane >> 4;

    const unsigned short* arow = A + (size_t)(m0 + r) * K;   // rows m0+r, +16*K
    const unsigned short* brow = B + (size_t)(n0 + r) * K;   // cols n0+r, +t*16*K

    const v8f vzero = {0.f, 0.f, 0.f, 0.f, 0.f, 0.f, 0.f, 0.f};
    v8f acc[2][4];
#pragma unroll
    for (int i = 0; i < 2; ++i)
#pragma unroll
        for (int t = 0; t < 4; ++t) acc[i][t] = vzero;

    // Software pipeline: K is a multiple of 64 (1024 or 2048 here).
    FragSet s0, s1;
    load_set<K>(s0, arow, brow, 0, h);

    int k0 = 0;
#pragma unroll 1
    for (; k0 < K - 64; k0 += 64) {
        load_set<K>(s1, arow, brow, k0 + 32, h);
        mma_set(s0, acc);
        load_set<K>(s0, arow, brow, k0 + 64, h);
        mma_set(s1, acc);
    }
    // Final 64-K slice: K-64 already resident in s0, K-32 still to load.
    load_set<K>(s1, arow, brow, k0 + 32, h);
    mma_set(s0, acc);
    mma_set(s1, acc);

#pragma unroll
    for (int i = 0; i < 2; ++i) {
        const int rowb = m0 + i * 16 + h * 8;
#pragma unroll
        for (int t = 0; t < 4; ++t) {
            const int col = n0 + t * 16 + r;
            const float bv = bias ? bias[col] : 0.f;
#pragma unroll
            for (int j = 0; j < 8; ++j) {
                C[(size_t)(rowb + j) * N + col] = acc[i][t][j] + bv;
            }
        }
    }
}

// ---------------------------------------------------------------------------
// Causal depthwise conv (D_CONV=4) + SiLU.
//   u[b,l,d] = silu( sum_{j=0..3} cw[d,j] * xz[b, l+j-3, d] )   (zero-padded)
// xz is (B*L, 2*D) row-major; channel part is columns [0, D).
// ---------------------------------------------------------------------------
__global__ void k_conv_silu(const float* __restrict__ xz,
                            const float* __restrict__ cw,
                            float* __restrict__ u,
                            int Bb, int L, int D) {
    int i = blockIdx.x * blockDim.x + threadIdx.x;
    int total = Bb * L * D;
    if (i >= total) return;
    int d = i % D;
    int m = i / D;
    int l = m % L;
    int b = m / L;
    const size_t stride = (size_t)(2 * D);
    float acc = 0.f;
#pragma unroll
    for (int j = 0; j < 4; ++j) {
        int ls = l + j - 3;
        if (ls >= 0)
            acc += cw[d * 4 + j] * xz[((size_t)(b * L + ls)) * stride + d];
    }
    u[i] = acc / (1.f + __expf(-acc));  // silu(acc) = acc*sigmoid(acc)
}

// ---------------------------------------------------------------------------
// x_proj: x_dbl[m,e] = sum_d u[m,d] * w[e,d],  e in [0,17). One wave / output.
// ---------------------------------------------------------------------------
__global__ void k_xproj(const float* __restrict__ u,
                        const float* __restrict__ w,
                        float* __restrict__ xdbl,
                        int M, int K, int E) {
    int wave = blockIdx.x * (blockDim.x >> 5) + (threadIdx.x >> 5);
    int lane = threadIdx.x & 31;
    int m = wave / E;
    int e = wave - m * E;
    if (m >= M) return;
    const float* ur = u + (size_t)m * K;
    const float* wr = w + (size_t)e * K;
    float s = 0.f;
    for (int k = lane; k < K; k += 32) s += ur[k] * wr[k];
#pragma unroll
    for (int off = 16; off > 0; off >>= 1) s += __shfl_down(s, off, 32);
    if (lane == 0) xdbl[(size_t)m * E + e] = s;
}

// ---------------------------------------------------------------------------
// delta[m,d] = softplus(x_dbl[m,0] * dt_w[d] + dt_b[d])
// ---------------------------------------------------------------------------
__global__ void k_delta(const float* __restrict__ xdbl,
                        const float* __restrict__ dtw,
                        const float* __restrict__ dtb,
                        float* __restrict__ delta,
                        int M, int D, int E) {
    int i = blockIdx.x * blockDim.x + threadIdx.x;
    if (i >= M * D) return;
    int d = i % D;
    int m = i / D;
    float v = xdbl[(size_t)m * E] * dtw[d] + dtb[d];
    delta[i] = (v > 20.f) ? v : log1pf(__expf(v));
}

// ---------------------------------------------------------------------------
// Selective scan: one lane per (b,d) channel, 16-state recurrence in regs.
//   h[n] <- exp(A[d,n]*delta)*h[n] + delta*Bt[n];  y = <h,Bt>
//   y_out = (y + u*Dp[d]) * silu(z);  stored as bf16 for the out_proj GEMM.
// ---------------------------------------------------------------------------
__global__ void k_scan(const float* __restrict__ delta,
                       const float* __restrict__ xdbl,
                       const float* __restrict__ xz,
                       const float* __restrict__ u,
                       const float* __restrict__ A_param,
                       const float* __restrict__ Dp,
                       unsigned short* __restrict__ ybf,
                       int Bb, int L, int D, int E) {
    int t = blockIdx.x * blockDim.x + threadIdx.x;
    if (t >= Bb * D) return;
    int b = t / D;
    int d = t - b * D;

    float a[16], h[16];
#pragma unroll
    for (int n = 0; n < 16; ++n) {
        a[n] = -__expf(A_param[(size_t)d * 16 + n]);
        h[n] = 0.f;
    }
    const float dpv = Dp[d];
    const size_t strideXZ = (size_t)(2 * D);

    for (int l = 0; l < L; ++l) {
        const size_t m = (size_t)b * L + l;
        const float dlt = delta[m * D + d];
        const float* Bt = xdbl + m * E + 1;   // 16 B-values, shared per (b,l)
        float y = 0.f;
#pragma unroll
        for (int n = 0; n < 16; ++n) {
            float bn = Bt[n];
            h[n] = __expf(a[n] * dlt) * h[n] + dlt * bn;
            y += h[n] * bn;
        }
        const float uv = u[m * D + d];
        const float zv = xz[m * strideXZ + D + d];
        const float sz = zv / (1.f + __expf(-zv));
        ybf[m * D + d] = f32_to_bf16_rne((y + uv * dpv) * sz);
    }
}

// ---------------------------------------------------------------------------
// Host-side orchestration
// ---------------------------------------------------------------------------
extern "C" void kernel_launch(void* const* d_in, const int* in_sizes, int n_in,
                              void* d_out, int out_size, void* d_ws, size_t ws_size,
                              hipStream_t stream) {
    (void)in_sizes; (void)n_in; (void)out_size; (void)ws_size;

    constexpr int Bb = 2, L = 1024, DM = 1024, DI = 2048, E = 17;
    constexpr int M = Bb * L;                  // 2048 rows (tokens)

    const float* x        = (const float*)d_in[0];   // (B,L,DM)
    const float* in_w     = (const float*)d_in[1];   // (2*DI, DM)
    const float* conv_w   = (const float*)d_in[2];   // (DI,1,4)
    const float* xproj_w  = (const float*)d_in[3];   // (E, DI)
    const float* dt_w     = (const float*)d_in[4];   // (DI,1)
    const float* dt_b     = (const float*)d_in[5];   // (DI,)
    const float* A_param  = (const float*)d_in[6];   // (DI,16)
    const float* D_param  = (const float*)d_in[7];   // (DI,)
    const float* out_w    = (const float*)d_in[8];   // (DM, DI)
    const float* out_b    = (const float*)d_in[9];   // (DM,)
    float* out = (float*)d_out;                      // (B,L,DM) f32

    // Workspace carve-up (256B aligned)
    char* base = (char*)d_ws;
    size_t off = 0;
    auto alloc = [&](size_t bytes) -> void* {
        off = (off + 255) & ~(size_t)255;
        void* p = base + off;
        off += bytes;
        return p;
    };
    unsigned short* xbf   = (unsigned short*)alloc((size_t)M * DM * 2);       // 4 MB
    unsigned short* wibf  = (unsigned short*)alloc((size_t)2 * DI * DM * 2);  // 8 MB
    unsigned short* wobf  = (unsigned short*)alloc((size_t)DM * DI * 2);      // 4 MB
    float*          xz    = (float*)alloc((size_t)M * 2 * DI * 4);            // 32 MB
    float*          u     = (float*)alloc((size_t)M * DI * 4);                // 16 MB
    float*          xdbl  = (float*)alloc((size_t)M * E * 4);
    float*          delta = (float*)alloc((size_t)M * DI * 4);                // 16 MB
    unsigned short* ybf   = (unsigned short*)alloc((size_t)M * DI * 2);       // 8 MB

    // 1) fp32 -> bf16 conversions
    {
        int n1 = M * DM;       k_f32_to_bf16<<<(n1 + 255) / 256, 256, 0, stream>>>(x, xbf, n1);
        int n2 = 2 * DI * DM;  k_f32_to_bf16<<<(n2 + 255) / 256, 256, 0, stream>>>(in_w, wibf, n2);
        int n3 = DM * DI;      k_f32_to_bf16<<<(n3 + 255) / 256, 256, 0, stream>>>(out_w, wobf, n3);
    }

    // 2) in_proj GEMM: xz(M x 4096) = xbf(M x 1024) * wibf(4096 x 1024)^T
    {
        int tiles  = (M / 32) * ((2 * DI) / 64);         // 4096 waves
        int blocks = tiles / 4;                          // 4 waves / block
        k_wmma_gemm_nt<DM><<<blocks, 128, 0, stream>>>(xbf, wibf, xz, nullptr,
                                                       M, 2 * DI);
    }

    // 3) depthwise causal conv + SiLU -> u (M x DI)
    {
        int n = M * DI;
        k_conv_silu<<<(n + 255) / 256, 256, 0, stream>>>(xz, conv_w, u, Bb, L, DI);
    }

    // 4) x_proj: xdbl(M x 17) = u * xproj_w^T (wave per output)
    {
        int waves  = M * E;
        int blocks = (waves + 7) / 8;                    // 8 waves / block
        k_xproj<<<blocks, 256, 0, stream>>>(u, xproj_w, xdbl, M, DI, E);
    }

    // 5) delta = softplus(dt_in * dt_w + dt_b)
    {
        int n = M * DI;
        k_delta<<<(n + 255) / 256, 256, 0, stream>>>(xdbl, dt_w, dt_b, delta, M, DI, E);
    }

    // 6) selective scan (+ skip, gate, bf16 cast)
    {
        int n = Bb * DI;                                 // 4096 channels
        k_scan<<<(n + 255) / 256, 256, 0, stream>>>(delta, xdbl, xz, u,
                                                    A_param, D_param, ybf,
                                                    Bb, L, DI, E);
    }

    // 7) out_proj GEMM: out(M x 1024) = ybf(M x 2048) * wobf(1024 x 2048)^T + out_b
    {
        int tiles  = (M / 32) * (DM / 64);               // 1024 waves
        int blocks = tiles / 4;
        k_wmma_gemm_nt<DI><<<blocks, 128, 0, stream>>>(ybf, wobf, out, out_b,
                                                       M, DM);
    }
}
